// RoPEMultiheadAttention_85083302133969
// MI455X (gfx1250) — matrix operand order repaired and checked by reading, not compiled, not run on hardware
//
#include <hip/hip_runtime.h>
#include <hip/hip_bf16.h>
#include <math.h>

#define D_MODEL 1024
#define NUM_HEADS 16
#define HEAD_DIM 64
#define SEQ 4096

typedef _Float16 h16;
typedef __attribute__((ext_vector_type(16))) _Float16 v16h;
typedef __attribute__((ext_vector_type(8)))  _Float16 v8h;
typedef __attribute__((ext_vector_type(8)))  float    v8f;
typedef __attribute__((ext_vector_type(4)))  float    v4f;

__device__ __forceinline__ v8f wmma16(v16h a, v16h b, v8f c) {
  // D = A(16x32 f16) * B(32x16 f16) + C(16x16 f32)
  return __builtin_amdgcn_wmma_f32_16x16x32_f16(false, a, false, b, (short)0, c,
                                                false, false);
}

// Async DMA: 16 bytes global -> LDS per lane (ASYNCcnt-tracked).
__device__ __forceinline__ void async_copy16(unsigned lds_off, const void* gptr) {
  asm volatile("global_load_async_to_lds_b128 %0, %1, off"
               :: "v"(lds_off), "v"((unsigned long long)gptr)
               : "memory");
}
__device__ __forceinline__ unsigned lds_addr_of(const void* p) {
  return (unsigned)(unsigned long long)p;   // generic LDS ptr: addr[31:0] = LDS offset
}

// A fragment (16x32 f16) from row-major f16, ld in elements.
// Lane l: row = row0 + (l&15); halves[0..7] = K k0+hi*8+i, halves[8..15] = +16.
__device__ __forceinline__ v16h afrag_f16(const h16* base, int ld, int row0,
                                          int k0, int lane) {
  const int m = lane & 15, hi = lane >> 4;
  const h16* p = base + (size_t)(row0 + m) * ld + k0 + hi * 8;
  union { v16h v; v8h h[2]; } u;
  u.h[0] = *(const v8h*)(p);
  u.h[1] = *(const v8h*)(p + 16);
  return u.v;
}

// A fragment (16x32) converted on the fly from fp32 row-major.
__device__ __forceinline__ v16h afrag_f32(const float* base, int ld, int row0,
                                          int k0, int lane) {
  const int m = lane & 15, hi = lane >> 4;
  const float* p = base + (size_t)(row0 + m) * ld + k0 + hi * 8;
  v4f f0 = *(const v4f*)(p);
  v4f f1 = *(const v4f*)(p + 4);
  v4f f2 = *(const v4f*)(p + 16);
  v4f f3 = *(const v4f*)(p + 20);
  v16h a;
#pragma unroll
  for (int i = 0; i < 4; ++i) {
    a[i]      = (h16)f0[i];
    a[4 + i]  = (h16)f1[i];
    a[8 + i]  = (h16)f2[i];
    a[12 + i] = (h16)f3[i];
  }
  return a;
}

// B fragment (32x16 f16) where column n is contiguous along k:
// B[k][n] = colbase[n*ldcol + k0 + k].  Lane l: col = l&15, halves[0..7] = k hi*8+i, halves[8..15] = +16.
__device__ __forceinline__ v16h bfrag_cols(const h16* colbase, int ldcol,
                                           int k0, int lane) {
  const int n = lane & 15, hi = lane >> 4;
  const h16* p = colbase + (size_t)n * ldcol + k0 + hi * 8;
  union { v16h v; v8h h[2]; } u;
  u.h[0] = *(const v8h*)(p);
  u.h[1] = *(const v8h*)(p + 16);
  return u.v;
}

// ---------------- Kernel 1: weights fp32 -> f16 (Wq,Wk,Wv,Wo) ---------------
__global__ void cvt_weights_kernel(const float* __restrict__ Wq,
                                   const float* __restrict__ Wk,
                                   const float* __restrict__ Wv,
                                   const float* __restrict__ Wo,
                                   h16* __restrict__ dst) {
  const int total = 4 * D_MODEL * D_MODEL;
  for (int i = blockIdx.x * blockDim.x + threadIdx.x; i < total;
       i += gridDim.x * blockDim.x) {
    const int which = i >> 20;
    const int off = i & ((1 << 20) - 1);
    const float* src = (which == 0) ? Wq : (which == 1) ? Wk
                        : (which == 2) ? Wv : Wo;
    dst[i] = (h16)src[off];
  }
}

// ------- Kernel 2: QKV projection GEMM + bias + RoPE (Q,K) / V-transpose ----
// grid = (NUM_HEADS, SEQ/128, 3), block = (32, 8). Wave owns a 16x64 tile.
// Double-buffered k-loop: fragments for k+32 load while k's WMMAs issue.
__global__ void qkv_rope_kernel(const float* __restrict__ xq,
                                const float* __restrict__ xk,
                                const float* __restrict__ xv,
                                const float* __restrict__ bq,
                                const float* __restrict__ bk,
                                const float* __restrict__ bv,
                                const h16* __restrict__ W16base,
                                h16* __restrict__ Qh, h16* __restrict__ Kh,
                                h16* __restrict__ Vt) {
  const int lane = threadIdx.x, wv = threadIdx.y;
  const int z = blockIdx.z;                       // 0=Q 1=K 2=V
  const int head = blockIdx.x;
  const int m0 = (blockIdx.y * 8 + wv) * 16;
  const int n0 = head * HEAD_DIM;
  const int hi = lane >> 4, ln = lane & 15;

  const float* X    = (z == 0) ? xq : (z == 1) ? xk : xv;
  const float* bias = (z == 0) ? bq : (z == 1) ? bk : bv;
  const h16* W16 = W16base + (size_t)z * D_MODEL * D_MODEL;

  v8f acc[4];
#pragma unroll
  for (int t = 0; t < 4; ++t) {
    const float b = bias[n0 + t * 16 + ln];
#pragma unroll
    for (int r = 0; r < 8; ++r) acc[t][r] = b;
  }

  v16h a_cur = afrag_f32(X, D_MODEL, m0, 0, lane);
  v16h b_cur[4];
#pragma unroll
  for (int t = 0; t < 4; ++t)
    b_cur[t] = bfrag_cols(W16 + (size_t)(n0 + t * 16) * D_MODEL, D_MODEL, 0, lane);

  for (int k0 = 0; k0 < D_MODEL; k0 += 32) {
    const int kn = (k0 + 32) & (D_MODEL - 1);        // wraps harmlessly on last
    v16h a_nxt = afrag_f32(X, D_MODEL, m0, kn, lane);
    v16h b_nxt[4];
#pragma unroll
    for (int t = 0; t < 4; ++t)
      b_nxt[t] = bfrag_cols(W16 + (size_t)(n0 + t * 16) * D_MODEL, D_MODEL, kn, lane);
#pragma unroll
    for (int t = 0; t < 4; ++t) acc[t] = wmma16(a_cur, b_cur[t], acc[t]);
    a_cur = a_nxt;
#pragma unroll
    for (int t = 0; t < 4; ++t) b_cur[t] = b_nxt[t];
  }

  if (z < 2) {
    // RoPE: out[d] = x[d]*cos - x[d+32]*sin ; out[d+32] = x[d+32]*cos + x[d]*sin
    h16* dst = ((z == 0) ? Qh : Kh) + (size_t)head * SEQ * HEAD_DIM;
#pragma unroll
    for (int r = 0; r < 8; ++r) {
      const int s = m0 + r + 8 * hi;
#pragma unroll
      for (int t = 0; t < 2; ++t) {
        const int j = t * 16 + ln;                       // 0..31
        const float ang = (float)s * __expf(-(float)j * 0.2878231366242557f);
        float sn, cs;
        __sincosf(ang, &sn, &cs);
        const float x = acc[t][r];
        const float y = acc[t + 2][r];
        dst[(size_t)s * HEAD_DIM + j]      = (h16)(x * cs - y * sn);
        dst[(size_t)s * HEAD_DIM + j + 32] = (h16)(y * cs + x * sn);
      }
    }
  } else {
    // V stored transposed per head: Vt[h][d][s]
    h16* dst = Vt + (size_t)head * HEAD_DIM * SEQ;
#pragma unroll
    for (int r = 0; r < 8; ++r) {
      const int s = m0 + r + 8 * hi;
#pragma unroll
      for (int t = 0; t < 4; ++t) {
        const int d = t * 16 + ln;
        dst[(size_t)d * SEQ + s] = (h16)acc[t][r];
      }
    }
  }
}

// ---- Kernel 3: flash attention. Block = 8 waves, one head, 128 q rows. ----
// K/V tiles (shared by all 8 waves) are DMA'd to LDS with
// global_load_async_to_lds_b128, double-buffered, overlapping the WMMAs.
__global__ void flash_attn_kernel(const h16* __restrict__ Qh,
                                  const h16* __restrict__ Kh,
                                  const h16* __restrict__ Vt,
                                  h16* __restrict__ Ctx) {
  __shared__ __align__(16) h16 Kbuf[2][32 * 64];   // [key][dim]      4KB each
  __shared__ __align__(16) h16 Vbuf[2][64 * 32];   // [dim][key]      4KB each
  __shared__ __align__(16) h16 Pbuf[8][16 * 32];   // per-wave P      1KB each
  const int lane = threadIdx.x, wv = threadIdx.y;
  const int tid = wv * 32 + lane;                  // 0..255
  const int head = blockIdx.x;
  const int m0 = (blockIdx.y * 8 + wv) * 16;
  const int hi = lane >> 4, ln = lane & 15;
  const float scale = 0.125f;                      // 1/sqrt(64)

  const h16* Qb = Qh + (size_t)head * SEQ * HEAD_DIM;
  const h16* Kb = Kh + (size_t)head * SEQ * HEAD_DIM;
  const h16* Vb = Vt + (size_t)head * HEAD_DIM * SEQ;

  // Cooperative async DMA of one 32-key step into LDS buffer `buf`.
  auto issue_tiles = [&](int buf, int j0) {
    {
      const int row = tid >> 3, seg = tid & 7;     // 32 rows x 8 x 16B = 4KB
      async_copy16(lds_addr_of(&Kbuf[buf][row * 64 + seg * 8]),
                   Kb + (size_t)(j0 + row) * HEAD_DIM + seg * 8);
    }
    {
      const int d = tid >> 2, seg = tid & 3;       // 64 dims x 4 x 16B = 4KB
      async_copy16(lds_addr_of(&Vbuf[buf][d * 32 + seg * 8]),
                   Vb + (size_t)d * SEQ + j0 + seg * 8);
    }
  };

  issue_tiles(0, 0);

  const v16h aq0 = afrag_f16(Qb, HEAD_DIM, m0, 0, lane);
  const v16h aq1 = afrag_f16(Qb, HEAD_DIM, m0, 32, lane);

  v8f ctx[4];
  float m_run[8], l_run[8];
#pragma unroll
  for (int t = 0; t < 4; ++t)
#pragma unroll
    for (int r = 0; r < 8; ++r) ctx[t][r] = 0.f;
#pragma unroll
  for (int r = 0; r < 8; ++r) { m_run[r] = -1e30f; l_run[r] = 0.f; }

  for (int j0 = 0; j0 < SEQ; j0 += 32) {
    const int cur = (j0 >> 5) & 1;
    asm volatile("s_wait_asynccnt 0x0" ::: "memory");  // my DMA done
    __syncthreads();                                   // everyone's DMA done
    if (j0 + 32 < SEQ) issue_tiles(cur ^ 1, j0 + 32);  // overlap next DMA

    const h16* Kl = &Kbuf[cur][0];
    const h16* Vl = &Vbuf[cur][0];

    v8f s0 = {}, s1 = {};
    {
      v16h b = bfrag_cols(Kl, HEAD_DIM, 0, lane);            // keys j0..+15
      s0 = wmma16(aq0, b, s0);
      b = bfrag_cols(Kl, HEAD_DIM, 32, lane);
      s0 = wmma16(aq1, b, s0);
      b = bfrag_cols(Kl + 16 * HEAD_DIM, HEAD_DIM, 0, lane); // keys +16..+31
      s1 = wmma16(aq0, b, s1);
      b = bfrag_cols(Kl + 16 * HEAD_DIM, HEAD_DIM, 32, lane);
      s1 = wmma16(aq1, b, s1);
    }
    // Online softmax. Row m = r + 8*hi lives in one 16-lane half at VGPR r.
#pragma unroll
    for (int r = 0; r < 8; ++r) {
      const float x0 = s0[r] * scale, x1 = s1[r] * scale;
      float mx = fmaxf(x0, x1);
      mx = fmaxf(mx, __shfl_xor(mx, 1, 32));
      mx = fmaxf(mx, __shfl_xor(mx, 2, 32));
      mx = fmaxf(mx, __shfl_xor(mx, 4, 32));
      mx = fmaxf(mx, __shfl_xor(mx, 8, 32));
      const float m_new = fmaxf(m_run[r], mx);
      const float p0 = __expf(x0 - m_new);
      const float p1 = __expf(x1 - m_new);
      float sum = p0 + p1;
      sum += __shfl_xor(sum, 1, 32);
      sum += __shfl_xor(sum, 2, 32);
      sum += __shfl_xor(sum, 4, 32);
      sum += __shfl_xor(sum, 8, 32);
      const float alpha = __expf(m_run[r] - m_new);
      l_run[r] = l_run[r] * alpha + sum;
      m_run[r] = m_new;
#pragma unroll
      for (int t = 0; t < 4; ++t) ctx[t][r] *= alpha;
      const int m = r + 8 * hi;
      Pbuf[wv][m * 32 + ln]      = (h16)p0;   // C-layout -> [m][k] in LDS
      Pbuf[wv][m * 32 + 16 + ln] = (h16)p1;
    }
    asm volatile("s_wait_dscnt 0" ::: "memory");
    // Re-read P as a 16x32 A fragment (cross-lane transpose via LDS).
    v16h pf;
    {
      const h16* pb = &Pbuf[wv][ln * 32];
      union { v16h v; v8h h[2]; } u;
      u.h[0] = *(const v8h*)(pb + hi * 8);
      u.h[1] = *(const v8h*)(pb + hi * 8 + 16);
      pf = u.v;
    }
#pragma unroll
    for (int t = 0; t < 4; ++t) {
      const v16h bvf = bfrag_cols(Vl + t * 16 * 32, 32, 0, lane);
      ctx[t] = wmma16(pf, bvf, ctx[t]);
    }
  }

#pragma unroll
  for (int r = 0; r < 8; ++r) {
    const float inv = 1.0f / l_run[r];
    const int s = m0 + r + 8 * hi;
#pragma unroll
    for (int t = 0; t < 4; ++t)
      Ctx[(size_t)s * D_MODEL + head * HEAD_DIM + t * 16 + ln] =
          (h16)(ctx[t][r] * inv);
  }
}

// --------------- Kernel 4: output projection  out = Ctx @ Wo^T + bo --------
__global__ void out_proj_kernel(const h16* __restrict__ Ctx,
                                const h16* __restrict__ Wo16,
                                const float* __restrict__ bo,
                                float* __restrict__ out) {
  const int lane = threadIdx.x, wv = threadIdx.y;
  const int n0 = blockIdx.x * 64;
  const int m0 = (blockIdx.y * 8 + wv) * 16;
  const int hi = lane >> 4, ln = lane & 15;

  v8f acc[4];
#pragma unroll
  for (int t = 0; t < 4; ++t) {
    const float b = bo[n0 + t * 16 + ln];
#pragma unroll
    for (int r = 0; r < 8; ++r) acc[t][r] = b;
  }

  v16h a_cur = afrag_f16(Ctx, D_MODEL, m0, 0, lane);
  v16h b_cur[4];
#pragma unroll
  for (int t = 0; t < 4; ++t)
    b_cur[t] = bfrag_cols(Wo16 + (size_t)(n0 + t * 16) * D_MODEL, D_MODEL, 0, lane);

  for (int k0 = 0; k0 < D_MODEL; k0 += 32) {
    const int kn = (k0 + 32) & (D_MODEL - 1);
    v16h a_nxt = afrag_f16(Ctx, D_MODEL, m0, kn, lane);
    v16h b_nxt[4];
#pragma unroll
    for (int t = 0; t < 4; ++t)
      b_nxt[t] = bfrag_cols(Wo16 + (size_t)(n0 + t * 16) * D_MODEL, D_MODEL, kn, lane);
#pragma unroll
    for (int t = 0; t < 4; ++t) acc[t] = wmma16(a_cur, b_cur[t], acc[t]);
    a_cur = a_nxt;
#pragma unroll
    for (int t = 0; t < 4; ++t) b_cur[t] = b_nxt[t];
  }

#pragma unroll
  for (int r = 0; r < 8; ++r) {
    const int s = m0 + r + 8 * hi;
#pragma unroll
    for (int t = 0; t < 4; ++t)
      out[(size_t)s * D_MODEL + n0 + t * 16 + ln] = acc[t][r];
  }
}

extern "C" void kernel_launch(void* const* d_in, const int* in_sizes, int n_in,
                              void* d_out, int out_size, void* d_ws,
                              size_t ws_size, hipStream_t stream) {
  const float* q  = (const float*)d_in[0];
  const float* k  = (const float*)d_in[1];
  const float* v  = (const float*)d_in[2];
  const float* Wq = (const float*)d_in[3];
  const float* bq = (const float*)d_in[4];
  const float* Wk = (const float*)d_in[5];
  const float* bk = (const float*)d_in[6];
  const float* Wv = (const float*)d_in[7];
  const float* bv = (const float*)d_in[8];
  const float* Wo = (const float*)d_in[9];
  const float* bo = (const float*)d_in[10];
  float* out = (float*)d_out;

  // Workspace layout (f16 elements): 4 weight mats | Qh | Kh | Vt | Ctx  (40 MB)
  h16* ws = (h16*)d_ws;
  const size_t MM = (size_t)D_MODEL * D_MODEL;               // 1M elems
  const size_t HS = (size_t)NUM_HEADS * SEQ * HEAD_DIM;      // 4M elems
  h16* W16 = ws;
  h16* Qh  = ws + 4 * MM;
  h16* Kh  = Qh + HS;
  h16* Vt  = Kh + HS;
  h16* Ctx = Vt + HS;

  hipLaunchKernelGGL(cvt_weights_kernel, dim3(2048), dim3(256), 0, stream,
                     Wq, Wk, Wv, Wo, W16);
  hipLaunchKernelGGL(qkv_rope_kernel, dim3(NUM_HEADS, SEQ / 128, 3),
                     dim3(32, 8), 0, stream,
                     q, k, v, bq, bk, bv, W16, Qh, Kh, Vt);
  hipLaunchKernelGGL(flash_attn_kernel, dim3(NUM_HEADS, SEQ / 128),
                     dim3(32, 8), 0, stream, Qh, Kh, Vt, Ctx);
  hipLaunchKernelGGL(out_proj_kernel, dim3(D_MODEL / 64, SEQ / 128),
                     dim3(32, 8), 0, stream, Ctx, W16 + 3 * MM, bo, out);
}
